// SNN_43559558316811
// MI455X (gfx1250) — compile-verified, matrix operand order checked by reading, not run on hardware
//
#include <hip/hip_runtime.h>

typedef __attribute__((ext_vector_type(16))) _Float16 v16h;
typedef __attribute__((ext_vector_type(8)))  _Float16 v8h;
typedef __attribute__((ext_vector_type(8)))  float    v8f;

#define ALPHA 0.9048374180359596f
#define BETA  0.8187307530779818f

enum { B_ = 128, D_ = 784, H1_ = 512, H2_ = 256, O_ = 10, T_ = 200, NWG = 16 };

// workspace layout (bytes)
enum : size_t {
  OFF_BARS = 0,                              // 2*T_*4 = 1600 -> pad to 2048
  OFF_Z1A  = 2048,                           // f16 [128][512]
  OFF_Z1B  = OFF_Z1A + (size_t)B_ * H1_ * 2,
  OFF_Z2A  = OFF_Z1B + (size_t)B_ * H1_ * 2, // f16 [128][256]
  OFF_Z2B  = OFF_Z2A + (size_t)B_ * H2_ * 2,
  OFF_IN0  = OFF_Z2B + (size_t)B_ * H2_ * 2, // f32 [128][512] = inputs @ W_in
  OFF_PVIN = OFF_IN0 + (size_t)B_ * H1_ * 4, // packed f16 fragments
  OFF_PWH  = OFF_PVIN + (size_t)H1_ * H1_ * 2,
  OFF_PVH  = OFF_PWH + (size_t)H1_ * H2_ * 2,
  OFF_PWO  = OFF_PVH + (size_t)H2_ * H2_ * 2,
  OFF_END  = OFF_PWO + (size_t)H2_ * 16 * 2
};

__device__ __forceinline__ v8f wmma16(v16h a, v16h b, v8f c) {
  return __builtin_amdgcn_wmma_f32_16x16x32_f16(false, a, false, b, (short)0, c,
                                                false, false);
}

__device__ __forceinline__ v16h cat8(v8h lo, v8h hi) {
  return __builtin_shufflevector(lo, hi, 0, 1, 2, 3, 4, 5, 6, 7, 8, 9, 10, 11,
                                 12, 13, 14, 15);
}

// A-fragment (16x32 f16, M x K) from a row-major [B][Hk] f16 spike matrix.
// lane L: m = mt*16 + (L&15); elems 0..7 -> k = kt*32 + 8*(L>>4) + r,
//         elems 8..15 -> k = kt*32 + 8*(2 + (L>>4)) + r   (two 16B loads)
__device__ __forceinline__ v16h loadA(const _Float16* Z, int Hk, int mRow,
                                      int kt, int lane) {
  const _Float16* p = Z + mRow * Hk + kt * 32 + ((lane >> 4) << 3);
  return cat8(*(const v8h*)p, *(const v8h*)(p + 16));
}

// B-fragment from pre-packed storage: tile*512 + lane*16, contiguous 32B/lane.
__device__ __forceinline__ v16h loadB(const _Float16* base, int tile, int lane) {
  const _Float16* p = base + tile * 512 + lane * 16;
  return cat8(*(const v8h*)p, *(const v8h*)(p + 8));
}

// Grid barrier: one fresh counter per (step,phase); counters zeroed by kZero.
__device__ __forceinline__ void gbar(unsigned* c) {
  __syncthreads();
  if (threadIdx.x == 0) {
    __hip_atomic_fetch_add(c, 1u, __ATOMIC_RELEASE, __HIP_MEMORY_SCOPE_AGENT);
    while (__hip_atomic_load(c, __ATOMIC_ACQUIRE, __HIP_MEMORY_SCOPE_AGENT) <
           (unsigned)NWG)
      __builtin_amdgcn_s_sleep(1);
  }
  __syncthreads();
}

__global__ void kZero(unsigned* p, int nwords) {
  for (int i = blockIdx.x * blockDim.x + threadIdx.x; i < nwords;
       i += gridDim.x * blockDim.x)
    p[i] = 0u;
}

// Pack a row-major f32 [K][Nsrc] weight matrix into f16 WMMA B-fragments.
// Layout: tile = nt*ktiles + kt; within tile: [lane][e] with
// k = kt*32 + (e&7) + 8*(2*(e>>3) + (lane>>4)), n = nt*16 + (lane&15).
__global__ void kPack(const float* __restrict__ src, _Float16* __restrict__ dst,
                      int ktiles, int ntiles, int Nsrc) {
  int total = ktiles * ntiles * 512;
  for (int i = blockIdx.x * blockDim.x + threadIdx.x; i < total;
       i += gridDim.x * blockDim.x) {
    int e = i & 15, lane = (i >> 4) & 31, tile = i >> 9;
    int kt = tile % ktiles, nt = tile / ktiles;
    int k = kt * 32 + (e & 7) + 8 * (2 * (e >> 3) + (lane >> 4));
    int n = nt * 16 + (lane & 15);
    float v = (n < Nsrc) ? src[k * Nsrc + n] : 0.0f;
    dst[i] = (_Float16)v;
  }
}

// In0 = inputs @ W_in (f32, one-time, only consumed at t=0)
__global__ void kIn0(const float* __restrict__ x, const float* __restrict__ W,
                     float* __restrict__ out) {
  int b = blockIdx.x, h = threadIdx.x;  // 128 blocks x 512 threads
  float acc = 0.0f;
  for (int d = 0; d < D_; ++d) acc = fmaf(x[b * D_ + d], W[d * H1_ + h], acc);
  out[b * H1_ + h] = acc;
}

// Persistent fused SNN kernel: 16 WGs x 256 threads (8 waves).
// WG g owns H1 cols [g*32, g*32+32) and H2 cols [g*16, g*16+16).
// Wave w owns batch rows [w*16, w*16+16). WGs 0..7 (wave 0) do the readout.
__global__ __launch_bounds__(256) void kSNN(
    const float* __restrict__ In0, _Float16* Z1A, _Float16* Z1B, _Float16* Z2A,
    _Float16* Z2B, const _Float16* __restrict__ pVin,
    const _Float16* __restrict__ pWh, const _Float16* __restrict__ pVh,
    const _Float16* __restrict__ pWo, unsigned* bars, float* __restrict__ out) {
  // 56KB LDS: V_in slice [0,16384), W_h slice [16384,24576), V_h [24576,28672)
  __shared__ __align__(32) _Float16 lds[28672];
  const int g = blockIdx.x, tid = threadIdx.x, lane = tid & 31, w = tid >> 5;

  {  // cooperative preload of this WG's packed weight slices (contiguous)
    const uint4* s1 = (const uint4*)(pVin + (size_t)g * 16384);
    const uint4* s2 = (const uint4*)(pWh + (size_t)g * 8192);
    const uint4* s3 = (const uint4*)(pVh + (size_t)g * 4096);
    uint4* d = (uint4*)lds;
#pragma unroll
    for (int i = 0; i < 8; i++) d[tid + i * 256] = s1[tid + i * 256];
#pragma unroll
    for (int i = 0; i < 4; i++) d[2048 + tid + i * 256] = s2[tid + i * 256];
#pragma unroll
    for (int i = 0; i < 2; i++) d[3072 + tid + i * 256] = s3[tid + i * 256];
  }
  __syncthreads();

  const int hi = lane >> 4, n = lane & 15;
  const int row0 = w * 16 + (hi << 3);  // + r -> batch row (C/D layout)
  const int aRow = w * 16 + n;          // A-fragment batch row
  v8f I1a = {}, I1b = {}, V1a = {}, V1b = {}, Z1a = {}, Z1b = {};
  v8f I2 = {}, V2 = {}, Z2 = {};
  v8f Ir = {}, Vr = {};

  for (int t = 0; t < T_; ++t) {
    _Float16* zr1 = (t & 1) ? Z1B : Z1A;
    _Float16* zw1 = (t & 1) ? Z1A : Z1B;
    _Float16* zr2 = (t & 1) ? Z2B : Z2A;
    _Float16* zw2 = (t & 1) ? Z2A : Z2B;

    // ---- Phase A: layer-1 current cur1 = (t==0 ? x@W_in : Z1@V_in) ----
    v8f c0 = {}, c1 = {};
    if (t == 0) {
#pragma unroll
      for (int r = 0; r < 8; r++) {
        c0[r] = In0[(row0 + r) * H1_ + g * 32 + n];
        c1[r] = In0[(row0 + r) * H1_ + g * 32 + 16 + n];
      }
    } else {
      for (int kt = 0; kt < 16; ++kt) {
        v16h a = loadA(zr1, H1_, aRow, kt, lane);
        v16h b0 = loadB(lds, kt, lane);
        v16h b1 = loadB(lds, 16 + kt, lane);
        c0 = wmma16(a, b0, c0);
        c1 = wmma16(a, b1, c1);
      }
    }
#pragma unroll
    for (int r = 0; r < 8; r++) {  // LIF update, layer 1 (old I in V update)
      float vn0 = (BETA * V1a[r] + I1a[r]) * (1.0f - Z1a[r]);
      float vn1 = (BETA * V1b[r] + I1b[r]) * (1.0f - Z1b[r]);
      I1a[r] = ALPHA * I1a[r] + c0[r];
      I1b[r] = ALPHA * I1b[r] + c1[r];
      float z0 = vn0 > 1.0f ? 1.0f : 0.0f;
      float z1 = vn1 > 1.0f ? 1.0f : 0.0f;
      V1a[r] = vn0; V1b[r] = vn1; Z1a[r] = z0; Z1b[r] = z1;
      zw1[(row0 + r) * H1_ + g * 32 + n] = (_Float16)z0;
      zw1[(row0 + r) * H1_ + g * 32 + 16 + n] = (_Float16)z1;
    }
    gbar(bars + 2 * t);  // all Z1_new visible

    // ---- Phase B: cur2 = z1_new@W_h + Z2@V_h ----
    v8f c2 = {};
    for (int kt = 0; kt < 16; ++kt) {
      v16h a = loadA(zw1, H1_, aRow, kt, lane);
      v16h b = loadB(lds + 16384, kt, lane);
      c2 = wmma16(a, b, c2);
    }
    for (int kt = 0; kt < 8; ++kt) {
      v16h a = loadA(zr2, H2_, aRow, kt, lane);
      v16h b = loadB(lds + 24576, kt, lane);
      c2 = wmma16(a, b, c2);
    }
#pragma unroll
    for (int r = 0; r < 8; r++) {  // LIF update, layer 2
      float vn = (BETA * V2[r] + I2[r]) * (1.0f - Z2[r]);
      I2[r] = ALPHA * I2[r] + c2[r];
      float z = vn > 1.0f ? 1.0f : 0.0f;
      V2[r] = vn; Z2[r] = z;
      zw2[(row0 + r) * H2_ + g * 16 + n] = (_Float16)z;
    }
    gbar(bars + 2 * t + 1);  // all Z2_new visible

    // ---- Phase C: readout (WG g<8 owns batch tile g; wave 0 only) ----
    if (g < 8 && w == 0) {
      v8f c3 = {};
      for (int kt = 0; kt < 8; ++kt) {
        v16h a = loadA(zw2, H2_, g * 16 + n, kt, lane);
        v16h b = loadB(pWo, kt, lane);
        c3 = wmma16(a, b, c3);
      }
#pragma unroll
      for (int r = 0; r < 8; r++) {
        float vn = BETA * Vr[r] + Ir[r];  // old I
        Ir[r] = ALPHA * Ir[r] + c3[r];
        Vr[r] = vn;
        int b = g * 16 + (hi << 3) + r;
        if (n < O_) out[(b * T_ + t) * O_ + n] = vn;
      }
    }
  }
}

extern "C" void kernel_launch(void* const* d_in, const int* in_sizes, int n_in,
                              void* d_out, int out_size, void* d_ws,
                              size_t ws_size, hipStream_t stream) {
  const float* x = (const float*)d_in[0];
  const float* Wi = (const float*)d_in[1];
  const float* Vi = (const float*)d_in[2];
  const float* Wh = (const float*)d_in[3];
  const float* Vh = (const float*)d_in[4];
  const float* Wo = (const float*)d_in[5];
  char* ws = (char*)d_ws;
  unsigned* bars = (unsigned*)(ws + OFF_BARS);
  _Float16* Z1A = (_Float16*)(ws + OFF_Z1A);
  _Float16* Z1B = (_Float16*)(ws + OFF_Z1B);
  _Float16* Z2A = (_Float16*)(ws + OFF_Z2A);
  _Float16* Z2B = (_Float16*)(ws + OFF_Z2B);
  float* In0 = (float*)(ws + OFF_IN0);
  _Float16* pVin = (_Float16*)(ws + OFF_PVIN);
  _Float16* pWh = (_Float16*)(ws + OFF_PWH);
  _Float16* pVh = (_Float16*)(ws + OFF_PVH);
  _Float16* pWo = (_Float16*)(ws + OFF_PWO);

  kZero<<<128, 256, 0, stream>>>((unsigned*)ws, (int)(OFF_IN0 / 4));
  kPack<<<256, 256, 0, stream>>>(Vi, pVin, 16, 32, 512);
  kPack<<<256, 256, 0, stream>>>(Wh, pWh, 16, 16, 256);
  kPack<<<128, 256, 0, stream>>>(Vh, pVh, 8, 16, 256);
  kPack<<<8, 256, 0, stream>>>(Wo, pWo, 8, 1, 10);
  kIn0<<<B_, 512, 0, stream>>>(x, Wi, In0);
  kSNN<<<NWG, 256, 0, stream>>>(In0, Z1A, Z1B, Z2A, Z2B, pVin, pWh, pVh, pWo,
                                bars, (float*)d_out);
}
